// SeRNN_FWXX_4054449127986
// MI455X (gfx1250) — matrix operand order repaired and checked by reading, not compiled, not run on hardware
//
#include <hip/hip_runtime.h>

typedef __attribute__((ext_vector_type(16))) __bf16 v16bf;
typedef __attribute__((ext_vector_type(8)))  float  v8f;

struct __align__(16) U128 { unsigned int d[4]; };
struct __align__(16) F4   { float v[4]; };

union Frag16 {
  v16bf v;
  unsigned short u[16];
  U128 q[2];
};

__device__ __forceinline__ unsigned short f2bf(float f) {
  // round-to-nearest-even float -> bf16 bit pattern
  unsigned int u = __float_as_uint(f);
  u += 0x7FFFu + ((u >> 16) & 1u);
  return (unsigned short)(u >> 16);
}

#define LOG2E  1.44269504088896f

// Branch-free fast tanh: 2/(1+2^(-2*log2e*x)) - 1. v_exp_f32 + v_rcp_f32,
// saturates correctly at +/-1 for large |x| (rcp(inf)=0).
__device__ __forceinline__ float fast_tanh(float x) {
  float e = __builtin_amdgcn_exp2f(-2.0f * LOG2E * x);
  return 2.0f * __builtin_amdgcn_rcpf(1.0f + e) - 1.0f;
}

#define BB 1024
#define TT 256
#define CC 6
#define HH 32
#define GG 128

// -----------------------------------------------------------------------------
// Kernel 1: grouped LSTM scan.
// grid = (B/16, C), block = 256 threads (8 wave32's).
// Block owns a 16-row batch tile of one channel-LSTM for all T steps.
// Wave n computes gate columns [16n,16n+16) with ONE v_wmma_f32_16x16x32_bf16
// per step (K=32 == full hidden dim). W_hh fragment stays in VGPRs all scan.
// Branch-free exp2/rcp activations; cell state in registers; x7 stream-out
// uses non-temporal stores so the 201MB stream doesn't thrash L2.
// -----------------------------------------------------------------------------
__global__ __launch_bounds__(256) void lstm_scan_kernel(
    const float* __restrict__ x,     // [B,T,C]
    const float* __restrict__ w_ih,  // [C,G]
    const float* __restrict__ w_hh,  // [C,G,H]
    const float* __restrict__ b_ih,  // [C,G]
    const float* __restrict__ b_hh,  // [C,G]
    float* __restrict__ x7)          // [B,T,C*H]
{
  __shared__ float          xls[TT * 16];  // staged x tile, TRANSPOSED [T][16]
  __shared__ float          gts[16 * GG];  // activated gates [16][128]
  __shared__ unsigned short hA [16 * HH];  // hidden state bf16 [16][32]

  const int tid  = threadIdx.x;
  const int wave = tid >> 5;
  const int lane = tid & 31;
  const int nl   = lane & 15;       // column (N) / row-for-A index
  const int hi   = lane >> 4;       // which half of K / M this lane covers
  const int b0   = blockIdx.x * 16; // batch tile origin
  const int c    = blockIdx.y;      // channel

  // Gate column this lane owns in the D fragment.
  const int   g      = wave * 16 + nl;
  const float wih_l  = w_ih[(size_t)c * GG + g];
  const float bias_l = b_ih[(size_t)c * GG + g] + b_hh[(size_t)c * GG + g];

  // Unified activation constants: act = A*rcp(1 + exp2(-K*v)) + B
  // sigmoid (i,f,o gates): A=1, K=log2e,   B=0
  // tanh    (cell gate)  : A=2, K=2*log2e, B=-1
  const int   kind = wave >> 1;     // 0:i 1:f 2:g(cell) 3:o  (wave-uniform)
  const float actA = (kind == 2) ? 2.0f : 1.0f;
  const float actK = (kind == 2) ? (2.0f * LOG2E) : LOG2E;
  const float actB = (kind == 2) ? -1.0f : 0.0f;

  // Loop-invariant B fragment: B[k][n] = w_hh[c, g, k]; lane holds k=16*hi+e.
  Frag16 bfrag;
  {
    const float* wrow = w_hh + ((size_t)c * GG + g) * HH + 16 * hi;
    #pragma unroll
    for (int e = 0; e < 16; ++e) bfrag.u[e] = f2bf(wrow[e]);
  }

  // Stage x tile into LDS transposed; zero h. Cell state -> registers.
  for (int idx = tid; idx < 16 * TT; idx += 256) {
    int t = idx >> 4;        // [T][16] layout
    int m = idx & 15;
    xls[idx] = x[((size_t)(b0 + m) * TT + t) * CC + c];
  }
  for (int idx = tid; idx < 16 * HH; idx += 256) hA[idx] = 0;

  // Per-thread persistent state-update bookkeeping (2 elements per thread).
  float  creg[2] = {0.0f, 0.0f};
  int    gidx[2], hidx[2];
  float* x7p[2];
  #pragma unroll
  for (int r = 0; r < 2; ++r) {
    int e   = tid + 256 * r;
    int m   = e >> 5, h = e & 31;
    gidx[r] = m * GG + h;
    hidx[r] = m * HH + h;
    x7p [r] = x7 + (size_t)(b0 + m) * TT * (CC * HH) + c * HH + h;
  }
  __syncthreads();

  for (int t = 0; t < TT; ++t) {
    // A fragment (16x32 bf16 h-tile) from LDS: lane row nl, k-halves by hi.
    Frag16 af;
    af.q[0] = *(const U128*)&hA[nl * HH + 16 * hi];
    af.q[1] = *(const U128*)&hA[nl * HH + 16 * hi + 8];

    // 8 x values for rows m = 8*hi .. 8*hi+7: two 16B LDS loads.
    float xv[8];
    *(F4*)&xv[0] = *(const F4*)&xls[t * 16 + 8 * hi];
    *(F4*)&xv[4] = *(const F4*)&xls[t * 16 + 8 * hi + 4];

    v8f acc = {};
    acc = __builtin_amdgcn_wmma_f32_16x16x32_bf16(
        false, af.v, false, bfrag.v, (short)0, acc, false, false);

    // Epilogue: + x*w_ih + bias, branch-free activation, spill to LDS.
    #pragma unroll
    for (int j = 0; j < 8; ++j) {
      int   m = j + 8 * hi;
      float v = acc[j] + xv[j] * wih_l + bias_l;
      float e = __builtin_amdgcn_exp2f(-actK * v);
      float a = actA * __builtin_amdgcn_rcpf(1.0f + e) + actB;
      gts[m * GG + g] = a;
    }
    __syncthreads();

    // State update: 512 elements over 256 threads, c in registers.
    #pragma unroll
    for (int r = 0; r < 2; ++r) {
      float iv = gts[gidx[r]];
      float fv = gts[gidx[r] + 32];
      float gv = gts[gidx[r] + 64];
      float ov = gts[gidx[r] + 96];
      float cn = fv * creg[r] + iv * gv;
      creg[r]  = cn;
      float hv = ov * fast_tanh(cn);
      hA[hidx[r]] = f2bf(hv);
      __builtin_nontemporal_store(hv, x7p[r]);   // NT: keep stream out of L2
      x7p[r] += CC * HH;                          // advance one time step
    }
    __syncthreads();
  }
}

// -----------------------------------------------------------------------------
// Kernel 2: MLP head on last-timestep features.
// grid = B/16, block = 128 (4 waves). fc1 (192->64): 6 WMMAs per wave;
// fc3 (64->64): 2 WMMAs per wave; fc2 (64->2): scalar.
// -----------------------------------------------------------------------------
__global__ __launch_bounds__(128) void head_kernel(
    const float* __restrict__ x7,   // [B,T,192]
    const float* __restrict__ w1, const float* __restrict__ b1,
    const float* __restrict__ w3, const float* __restrict__ b3,
    const float* __restrict__ w2, const float* __restrict__ b2,
    float* __restrict__ out)        // [B,2]
{
  __shared__ unsigned short a1[16 * 192]; // last-step features, bf16
  __shared__ unsigned short h1[16 * 64];  // relu(fc1), bf16
  __shared__ float          h2[16 * 64];  // relu(fc3), f32

  const int tid  = threadIdx.x;
  const int wave = tid >> 5;
  const int lane = tid & 31;
  const int nl   = lane & 15;
  const int hi   = lane >> 4;
  const int b0   = blockIdx.x * 16;
  const int col  = wave * 16 + nl;

  __builtin_prefetch(w1, 0, 1);

  for (int idx = tid; idx < 16 * 192; idx += 128) {
    int m = idx / 192, k = idx % 192;
    a1[idx] = f2bf(x7[((size_t)(b0 + m) * TT + (TT - 1)) * 192 + k]);
  }
  __syncthreads();

  // fc1: K = 192 -> 6 WMMAs
  v8f acc = {};
  #pragma unroll
  for (int kk = 0; kk < 6; ++kk) {
    Frag16 af, bfr;
    af.q[0] = *(const U128*)&a1[nl * 192 + 32 * kk + 16 * hi];
    af.q[1] = *(const U128*)&a1[nl * 192 + 32 * kk + 16 * hi + 8];
    const float* wrow = w1 + (size_t)col * 192 + 32 * kk + 16 * hi;
    #pragma unroll
    for (int e = 0; e < 16; ++e) bfr.u[e] = f2bf(wrow[e]);
    acc = __builtin_amdgcn_wmma_f32_16x16x32_bf16(
        false, af.v, false, bfr.v, (short)0, acc, false, false);
  }
  {
    float bb = b1[col];
    #pragma unroll
    for (int j = 0; j < 8; ++j) {
      int   m = j + 8 * hi;
      float v = acc[j] + bb;
      h1[m * 64 + col] = f2bf(v > 0.0f ? v : 0.0f);
    }
  }
  __syncthreads();

  // fc3: K = 64 -> 2 WMMAs
  v8f acc3 = {};
  #pragma unroll
  for (int kk = 0; kk < 2; ++kk) {
    Frag16 af, bfr;
    af.q[0] = *(const U128*)&h1[nl * 64 + 32 * kk + 16 * hi];
    af.q[1] = *(const U128*)&h1[nl * 64 + 32 * kk + 16 * hi + 8];
    const float* wrow = w3 + (size_t)col * 64 + 32 * kk + 16 * hi;
    #pragma unroll
    for (int e = 0; e < 16; ++e) bfr.u[e] = f2bf(wrow[e]);
    acc3 = __builtin_amdgcn_wmma_f32_16x16x32_bf16(
        false, af.v, false, bfr.v, (short)0, acc3, false, false);
  }
  {
    float bb = b3[col];
    #pragma unroll
    for (int j = 0; j < 8; ++j) {
      int   m = j + 8 * hi;
      float v = acc3[j] + bb;
      h2[m * 64 + col] = v > 0.0f ? v : 0.0f;
    }
  }
  __syncthreads();

  // fc2: 2 outputs per batch row, scalar dot over 64.
  if (tid < 32) {
    int   m = tid >> 1, o = tid & 1;
    float s = b2[o];
    #pragma unroll 8
    for (int k = 0; k < 64; ++k) s += h2[m * 64 + k] * w2[o * 64 + k];
    out[(size_t)(b0 + m) * 2 + o] = s;
  }
}

extern "C" void kernel_launch(void* const* d_in, const int* in_sizes, int n_in,
                              void* d_out, int out_size, void* d_ws, size_t ws_size,
                              hipStream_t stream) {
  const float* x    = (const float*)d_in[0];
  const float* w_ih = (const float*)d_in[1];
  const float* w_hh = (const float*)d_in[2];
  const float* b_ih = (const float*)d_in[3];
  const float* b_hh = (const float*)d_in[4];
  const float* w1   = (const float*)d_in[5];
  const float* b1   = (const float*)d_in[6];
  const float* w3   = (const float*)d_in[7];
  const float* b3   = (const float*)d_in[8];
  const float* w2   = (const float*)d_in[9];
  const float* b2   = (const float*)d_in[10];

  float* out = (float*)d_out;                 // [B,2] first
  float* x7  = out + (size_t)BB * 2;          // then [B,T,192]

  dim3 g1(BB / 16, CC);
  lstm_scan_kernel<<<g1, 256, 0, stream>>>(x, w_ih, w_hh, b_ih, b_hh, x7);
  head_kernel<<<BB / 16, 128, 0, stream>>>(x7, w1, b1, w3, b3, w2, b2, out);
}
